// Net_77498389889242
// MI455X (gfx1250) — matrix operand order repaired and checked by reading, not compiled, not run on hardware
//
#include <hip/hip_runtime.h>

#define B_   512
#define T_   1024
#define I_   64
#define H_   128
#define O_   16
#define NT_  5
#define HS   136           // h row stride in bf16 elems (272 B: 16B-aligned, bank-skewed)
#define HBUF (16 * HS)

typedef __attribute__((ext_vector_type(16))) __bf16 v16bf;
typedef __attribute__((ext_vector_type(8)))  float  v8f;
typedef __attribute__((ext_vector_type(4)))  float  f4;

union BF16x16 { v16bf v; __bf16 b[16]; unsigned short u[16]; };

// ---- hardware tanh (CDNA5 TRANS32 op); sigmoid via tanh: no divides/exp ----
__device__ __forceinline__ float tanh_hw(float x) {
#if __has_builtin(__builtin_amdgcn_tanhf)
    return __builtin_amdgcn_tanhf(x);
#else
    float r;
    asm("v_tanh_f32 %0, %1\n\tv_nop" : "=v"(r) : "v"(x));   // v_nop covers TRANS RAW hazard
    return r;
#endif
}
__device__ __forceinline__ float sigm_hw(float x) {
    return __builtin_fmaf(0.5f, tanh_hw(0.5f * x), 0.5f);
}
__device__ __forceinline__ unsigned short bfbits(float f) {
    union { __bf16 b; unsigned short u; } cv;
    cv.b = (__bf16)f;               // native v_cvt bf16 on gfx1250
    return cv.u;
}

// ---- split workgroup barrier (signal after LDS stores, wait before LDS reads) ----
#if __has_builtin(__builtin_amdgcn_s_barrier_signal) && \
    __has_builtin(__builtin_amdgcn_s_barrier_wait)   && \
    __has_builtin(__builtin_amdgcn_s_wait_dscnt)
#define SPLIT_BAR 1
#endif
__device__ __forceinline__ void bar_signal() {
#ifdef SPLIT_BAR
    __builtin_amdgcn_s_wait_dscnt(0);           // h stores visible in LDS
    __builtin_amdgcn_s_barrier_signal(-1);
#else
    __syncthreads();
#endif
}
__device__ __forceinline__ void bar_wait() {
#ifdef SPLIT_BAR
    __builtin_amdgcn_s_barrier_wait(-1);
#endif
}

// A-fragment (16x32 bf16) from an LDS bf16 row: lane = M, K pattern per ISA
__device__ __forceinline__ v16bf ldsAfrag(const unsigned short* row, int k0, int h8) {
    BF16x16 a;
    const unsigned short* p0 = row + k0 + h8;
    const unsigned short* p1 = row + k0 + 16 + h8;
    #pragma unroll
    for (int i = 0; i < 8; ++i) { a.u[i] = p0[i]; a.u[8 + i] = p1[i]; }
    return a.v;
}
// Issue NT loads of one step's raw x slice (32 floats, 8x b128)
__device__ __forceinline__ void xLoadRaw(f4* q, const float* xpRun) {
    #pragma unroll
    for (int j = 0; j < 8; ++j)     // runs of 8 floats at 0,16,32,48 (+h8 folded in)
        q[j] = __builtin_nontemporal_load((const f4*)(xpRun + (j >> 1) * 16 + (j & 1) * 4));
}
// Convert raw x into the two A-fragments (K 0..31 and 32..63)
__device__ __forceinline__ void xCvt(const f4* q, BF16x16& a0, BF16x16& a1) {
    #pragma unroll
    for (int e = 0; e < 16; ++e) {
        a0.b[e] = (__bf16)q[(e >> 2)][e & 3];
        a1.b[e] = (__bf16)q[4 + (e >> 2)][e & 3];
    }
}

#define WMMA_BF16(A, Bv, C) \
    __builtin_amdgcn_wmma_f32_16x16x32_bf16(false, (A), false, (Bv), (short)0, (C), false, false)

__global__ __launch_bounds__(256)
void ticker_lstm_fused(const float* __restrict__ x,
                       const int*   __restrict__ seq_lens,
                       const float* __restrict__ W_ci,
                       const float* __restrict__ b_ci,
                       const float* __restrict__ W_ig,
                       const float* __restrict__ b_ig,
                       const float* __restrict__ W_og,
                       const float* __restrict__ b_og,
                       const float* __restrict__ b_ci_t,
                       const float* __restrict__ b_ig_t,
                       const float* __restrict__ b_og_t,
                       const float* __restrict__ W_fc,
                       const float* __restrict__ b_fc,
                       float* __restrict__ out)
{
    __shared__ unsigned short hbuf[2 * HBUF];   // ping-pong h state, bf16 [16][HS]

    const int tid   = threadIdx.x;
    const int wave  = tid >> 5;
    const int lane  = tid & 31;
    const int laneN = lane & 15;
    const int hi    = lane >> 4;
    const int h8    = hi * 8;                   // A-frag K base shift
    const int h16   = hi * 16;                  // unused; B frags also shift K by h8
    const int col   = wave * 16 + laneN;        // owned H column (C/D: lane = N)
    const int mBase = blockIdx.x * 16;          // batch-tile base row

    for (int i = tid; i < HBUF; i += 256) hbuf[i] = 0;   // zero buffer 0 only

    const float bci = b_ci[col], big = b_ig[col], bog = b_og[col];

    int lens[8];
    #pragma unroll
    for (int r = 0; r < 8; ++r) lens[r] = seq_lens[mBase + r + h8];

    // Resident B-fragments for the scan only (Bfc & ticker biases loaded later).
    BF16x16 Bci[2], Big[4], Bog[4];
    #pragma unroll
    for (int f = 0; f < 2; ++f)
        #pragma unroll
        for (int e = 0; e < 16; ++e)
            Bci[f].b[e] = (__bf16)W_ci[(f * 32 + h8 + (e & 7) + ((e >> 3) << 4)) * H_ + col];
    #pragma unroll
    for (int f = 0; f < 4; ++f)
        #pragma unroll
        for (int e = 0; e < 16; ++e) {
            const int k = f * 32 + h8 + (e & 7) + ((e >> 3) << 4);
            Big[f].b[e] = (__bf16)W_ig[k * H_ + col];
            Bog[f].b[e] = (__bf16)W_og[k * H_ + col];
        }

    v8f c_reg = {}, h_reg = {};
    const float* xrow = x + (size_t)(mBase + laneN) * T_ * I_ + h8;  // h8 folded into run base

    // Prime the bf16 x-fragment pipeline with t=0; signal barrier round 0
    BF16x16 xf0, xf1;
    {
        f4 q[8];
        xLoadRaw(q, xrow);
        xCvt(q, xf0, xf1);
    }
    bar_signal();

    // ---------------- main scan: pipeline depth 1, ~16 persistent pipeline regs ----
    auto mainStep = [&](int t, int pr) {
        // consume this step's x fragments first
        v8f accX = {};
        accX = WMMA_BF16(xf0.v, Bci[0].v, accX);
        accX = WMMA_BF16(xf1.v, Bci[1].v, accX);

        // issue next step's raw x loads (waited on only at end of this step)
        f4 q[8];
        if (t + 1 < T_) xLoadRaw(q, xrow + (size_t)(t + 1) * I_);
        if (t + 2 < T_) __builtin_prefetch(xrow + (size_t)(t + 2) * I_, 0, 1);

        bar_wait();                                     // prior step's h visible
        const unsigned short* hrow = hbuf + pr * HBUF + laneN * HS;
        v16bf ha0 = ldsAfrag(hrow, 0,  h8);
        v16bf ha1 = ldsAfrag(hrow, 32, h8);
        v16bf ha2 = ldsAfrag(hrow, 64, h8);
        v16bf ha3 = ldsAfrag(hrow, 96, h8);
        v8f accI = {}, accG = {};
        accI = WMMA_BF16(ha0, Big[0].v, accI);
        accI = WMMA_BF16(ha1, Big[1].v, accI);
        accI = WMMA_BF16(ha2, Big[2].v, accI);
        accI = WMMA_BF16(ha3, Big[3].v, accI);
        accG = WMMA_BF16(ha0, Bog[0].v, accG);
        accG = WMMA_BF16(ha1, Bog[1].v, accG);
        accG = WMMA_BF16(ha2, Bog[2].v, accG);
        accG = WMMA_BF16(ha3, Bog[3].v, accG);

        #pragma unroll
        for (int r = 0; r < 8; ++r) {
            const float ci = tanh_hw(accX[r] + bci);
            const float ig = sigm_hw(accI[r] + big);
            const float og = sigm_hw(accG[r] + bog);
            const float cn = c_reg[r] + ig * ci;
            const float hn = og * cn;
            const bool  m  = (t < lens[r]);             // updates freeze at seq_len
            c_reg[r] = m ? cn : c_reg[r];
            h_reg[r] = m ? hn : h_reg[r];
        }

        // refill the x-fragment pipeline (loads have had the whole step to land)
        if (t + 1 < T_) xCvt(q, xf0, xf1);

        unsigned short* nb = hbuf + (pr ^ 1) * HBUF;    // write next-step buffer
        #pragma unroll
        for (int r = 0; r < 8; ++r)                     // D layout: VGPR r -> row r + h8
            nb[(r + h8) * HS + col] = bfbits(h_reg[r]);
        bar_signal();
    };

    #pragma unroll 1
    for (int t = 0; t < T_; t += 2) {                   // unroll x2: parity is constant
        mainStep(t,     0);
        mainStep(t + 1, 1);
    }
    // T_ is even -> final state is in parity 0 buffer

    // ---------------- ticker loop: 5 unmasked steps ----------------
    const float bcit = b_ci_t[col], bigt = b_ig_t[col], bogt = b_og_t[col];
    const float ci_t = tanh_hw(bci + bcit);
    int par = 0;
    #pragma unroll 1
    for (int it = 0; it < NT_; ++it) {
        bar_wait();
        const unsigned short* hrow = hbuf + par * HBUF + laneN * HS;
        v16bf ha0 = ldsAfrag(hrow, 0,  h8);
        v16bf ha1 = ldsAfrag(hrow, 32, h8);
        v16bf ha2 = ldsAfrag(hrow, 64, h8);
        v16bf ha3 = ldsAfrag(hrow, 96, h8);
        v8f accI = {}, accG = {};
        accI = WMMA_BF16(ha0, Big[0].v, accI);
        accI = WMMA_BF16(ha1, Big[1].v, accI);
        accI = WMMA_BF16(ha2, Big[2].v, accI);
        accI = WMMA_BF16(ha3, Big[3].v, accI);
        accG = WMMA_BF16(ha0, Bog[0].v, accG);
        accG = WMMA_BF16(ha1, Bog[1].v, accG);
        accG = WMMA_BF16(ha2, Bog[2].v, accG);
        accG = WMMA_BF16(ha3, Bog[3].v, accG);
        #pragma unroll
        for (int r = 0; r < 8; ++r) {
            const float ig = sigm_hw(accI[r] + big + bigt);
            const float og = sigm_hw(accG[r] + bog + bogt);
            c_reg[r] += ig * ci_t;
            h_reg[r]  = og * c_reg[r];
        }
        unsigned short* nb = hbuf + (par ^ 1) * HBUF;
        #pragma unroll
        for (int r = 0; r < 8; ++r)
            nb[(r + h8) * HS + col] = bfbits(h_reg[r]);
        bar_signal();
        par ^= 1;
    }

    // ---------------- output projection: h(16x128) @ W_fc(128x16) ----------------
    {
        // W_fc fragments loaded only now (kept out of the hot loop's registers)
        BF16x16 Bfc[4];
        #pragma unroll
        for (int f = 0; f < 4; ++f)
            #pragma unroll
            for (int e = 0; e < 16; ++e)
                Bfc[f].b[e] = (__bf16)W_fc[(f * 32 + h8 + (e & 7) + ((e >> 3) << 4)) * O_ + laneN];

        bar_wait();
        const unsigned short* hrow = hbuf + par * HBUF + laneN * HS;
        v16bf ha0 = ldsAfrag(hrow, 0,  h8);
        v16bf ha1 = ldsAfrag(hrow, 32, h8);
        v16bf ha2 = ldsAfrag(hrow, 64, h8);
        v16bf ha3 = ldsAfrag(hrow, 96, h8);
        v8f accF = {};
        accF = WMMA_BF16(ha0, Bfc[0].v, accF);
        accF = WMMA_BF16(ha1, Bfc[1].v, accF);
        accF = WMMA_BF16(ha2, Bfc[2].v, accF);
        accF = WMMA_BF16(ha3, Bfc[3].v, accF);
        if (wave == 0) {                                 // O=16 = exactly one N tile
            const float bf = b_fc[laneN];
            #pragma unroll
            for (int r = 0; r < 8; ++r)
                out[(size_t)(mBase + r + h8) * O_ + laneN] = accF[r] + bf;
        }
    }
}

extern "C" void kernel_launch(void* const* d_in, const int* in_sizes, int n_in,
                              void* d_out, int out_size, void* d_ws, size_t ws_size,
                              hipStream_t stream) {
    const float* x      = (const float*)d_in[0];
    const int*   lens   = (const int*)  d_in[1];
    const float* W_ci   = (const float*)d_in[2];
    const float* b_ci   = (const float*)d_in[3];
    const float* W_ig   = (const float*)d_in[4];
    const float* b_ig   = (const float*)d_in[5];
    const float* W_og   = (const float*)d_in[6];
    const float* b_og   = (const float*)d_in[7];
    const float* b_ci_t = (const float*)d_in[8];
    const float* b_ig_t = (const float*)d_in[9];
    const float* b_og_t = (const float*)d_in[10];
    const float* W_fc   = (const float*)d_in[11];
    const float* b_fc   = (const float*)d_in[12];
    float* out = (float*)d_out;

    ticker_lstm_fused<<<B_ / 16, 256, 0, stream>>>(
        x, lens, W_ci, b_ci, W_ig, b_ig, W_og, b_og,
        b_ci_t, b_ig_t, b_og_t, W_fc, b_fc, out);
}
